// BoxAwareXCorr_18098992185420
// MI455X (gfx1250) — matrix operand order repaired and checked by reading, not compiled, not run on hardware
//
#include <hip/hip_runtime.h>
#include <math.h>

// ---------------- problem constants ----------------
#define BB 4
#define MM 512
#define NN 1024
#define FF 256
#define HH 256
#define K2C 4      // K2
#define KCC 16     // KC
#define R5 (BB*NN*5)   // 20480 rows (group x 5 slots)
#define GN (BB*NN)     // 4096 groups / points

// ---------------- WMMA types ----------------
typedef __attribute__((ext_vector_type(16))) __bf16 v16bf;
typedef __attribute__((ext_vector_type(8)))  float  v8f;

union Frag {
  v16bf v;
  uint4 q[2];
};

__device__ __forceinline__ unsigned short f2bf(float f) {
  unsigned int u = __float_as_uint(f);
  unsigned int r = u + 0x7FFFu + ((u >> 16) & 1u);   // round-to-nearest-even
  return (unsigned short)(r >> 16);
}
__device__ __forceinline__ float bf2f(unsigned short h) {
  return __uint_as_float(((unsigned int)h) << 16);
}

// ============================================================
// Generic bf16 WMMA GEMM:  out = epi( A[M x Kpad] * W[O x Kpad]^T )
// A row-major bf16 (K padded to mult of 32, zero-filled),
// W packed row-major bf16 [O x Kpad] (i.e. out_o = sum_k W[o,k]*A[m,k]).
// Epilogue: v = relu?( scale[o]*acc + bias[o] ) + resid[m,o]
//
// Block = 8 waves sharing one 16-column tile: the 16 x Kpad weight strip is
// staged once into LDS (dynamic, <=32KB) with cooperative b128 copies, then
// each wave streams its A fragments from global (b128, WGP-scope prefetch)
// and B fragments from LDS (ds_load_b128) into v_wmma_f32_16x16x32_bf16.
// Fragment patterns follow the CDNA5 ISA VGPR layouts:
//   A 16x32 bf16: lane m=L&15, half h=L>>4: dwords kb/2+4h+{0..3}, kb/2+8+4h+{0..3}
//   B 32x16 bf16: lane n=L&15: dwords kb/2+4h+{0..3}, kb/2+8+4h+{0..3}
//   C 16x16 f32 : vgpr r -> m = r + 8h, n = L&15
// ============================================================
__global__ void __launch_bounds__(256) gemm_bf16_kernel(
    const unsigned short* __restrict__ A,
    const unsigned short* __restrict__ W,
    float* __restrict__ outF,
    unsigned short* __restrict__ outH,
    const float* __restrict__ scale,
    const float* __restrict__ bias,
    const float* __restrict__ resid,
    int Mrows, int Ocols, int Kpad, int relu)
{
  extern __shared__ uint4 sW[];                 // 16 rows x Kpad bf16
  const int lane = threadIdx.x & 31;
  const int wave = threadIdx.x >> 5;
  const int n0 = blockIdx.y << 4;

  // cooperative stage of the 16 x Kpad weight strip (contiguous in W)
  {
    const uint4* Wg = (const uint4*)(W + (size_t)n0 * Kpad);
    const int totq = (Kpad << 4) >> 3;          // 16*Kpad ushorts / 8 per uint4
    for (int i = threadIdx.x; i < totq; i += 256) sW[i] = Wg[i];
  }
  __syncthreads();

  const int rowTile = blockIdx.x * 8 + wave;
  if (rowTile * 16 >= Mrows) return;            // after the only barrier
  const int m0 = rowTile << 4;
  const int h   = lane >> 4;
  const int l16 = lane & 15;
  const uint4* Arow4 = (const uint4*)(A + (size_t)(m0 + l16) * Kpad);
  const uint4* Wrow4 = sW + (size_t)l16 * (Kpad >> 3);
  v8f acc = {0.f, 0.f, 0.f, 0.f, 0.f, 0.f, 0.f, 0.f};
  const int ksteps = Kpad >> 5;
  for (int ks = 0; ks < ksteps; ++ks) {
    const int q0 = ks << 2;                     // uint4 index of k-block start
    Frag a, b;
    a.q[0] = Arow4[q0 + h];
    a.q[1] = Arow4[q0 + 2 + h];
    b.q[0] = Wrow4[q0 + h];                     // ds_load_b128
    b.q[1] = Wrow4[q0 + 2 + h];
    if (ks + 1 < ksteps) {
      __builtin_prefetch((const void*)(Arow4 + q0 + 4), 0, 3);
    }
    acc = __builtin_amdgcn_wmma_f32_16x16x32_bf16(
        false, a.v, false, b.v, (short)0, acc, false, false);
  }
  const int o = n0 + l16;
  const float sc  = scale ? scale[o] : 1.0f;
  const float bsv = bias  ? bias[o]  : 0.0f;
#pragma unroll
  for (int r = 0; r < 8; ++r) {
    const int m = m0 + r + (h << 3);
    float v = sc * acc[r] + bsv;
    if (relu) v = fmaxf(v, 0.0f);
    const size_t idx = (size_t)m * Ocols + o;
    if (resid) v += resid[idx];
    if (outF) outF[idx] = v;
    if (outH) outH[idx] = f2bf(v);
  }
}

// ---------------- weight packing f32 -> bf16 (K padded) ----------------
__global__ void pack_w_kernel(const float* __restrict__ W, unsigned short* __restrict__ out,
                              int O, int K, int Kpad, int total)
{
  int gid = blockIdx.x * blockDim.x + threadIdx.x;
  if (gid >= total) return;
  int o = gid / Kpad;
  int k = gid - o * Kpad;
  out[gid] = (k < K) ? f2bf(W[(size_t)o * K + k]) : (unsigned short)0;
}

// ---------------- spherical coords about centroid -> (B,3,Np) ----------------
__global__ void scf_kernel(const float* __restrict__ xyz, float* __restrict__ out, int Np)
{
  int b = blockIdx.x;
  int tid = threadIdx.x;
  __shared__ float sm0[256], sm1[256], sm2[256];
  const float* base = xyz + (size_t)b * Np * 3;
  float ax = 0.f, ay = 0.f, az = 0.f;
  for (int i = tid; i < Np; i += 256) {
    ax += base[i * 3 + 0]; ay += base[i * 3 + 1]; az += base[i * 3 + 2];
  }
  sm0[tid] = ax; sm1[tid] = ay; sm2[tid] = az;
  __syncthreads();
  for (int s = 128; s > 0; s >>= 1) {
    if (tid < s) { sm0[tid] += sm0[tid + s]; sm1[tid] += sm1[tid + s]; sm2[tid] += sm2[tid + s]; }
    __syncthreads();
  }
  float mx = sm0[0] / (float)Np, my = sm1[0] / (float)Np, mz = sm2[0] / (float)Np;
  float* ob = out + (size_t)b * 3 * Np;
  for (int i = tid; i < Np; i += 256) {
    float rx = base[i * 3 + 0] - mx;
    float ry = base[i * 3 + 1] - my;
    float rz = base[i * 3 + 2] - mz;
    float r = sqrtf(rx * rx + ry * ry + rz * rz + 1e-8f);
    float ct = fminf(1.f, fmaxf(-1.f, rz / r));
    ob[i]            = r;
    ob[Np + i]       = acosf(ct);
    ob[2 * Np + i]   = atan2f(ry, rx);
  }
}

// ---------------- per-point SCF descriptor: 16 sorted NN distances ----------------
__global__ void knn_desc_kernel(const float* __restrict__ xyz, float* __restrict__ desc,
                                int Np, int total)
{
  int gid = blockIdx.x * blockDim.x + threadIdx.x;
  if (gid >= total) return;
  int b = gid / Np, i = gid - b * Np;
  const float* base = xyz + (size_t)b * Np * 3;
  float px = base[i * 3], py = base[i * 3 + 1], pz = base[i * 3 + 2];
  float best[17];
#pragma unroll
  for (int k = 0; k < 17; ++k) best[k] = 3.0e38f;
  for (int j = 0; j < Np; ++j) {
    float dx = base[j * 3] - px, dy = base[j * 3 + 1] - py, dz = base[j * 3 + 2] - pz;
    float d2 = dx * dx + dy * dy + dz * dz;
    if (d2 < best[16]) {
      int p = 16;
      while (p > 0 && best[p - 1] > d2) { best[p] = best[p - 1]; --p; }
      best[p] = d2;
    }
  }
  for (int k = 0; k < 16; ++k) desc[(size_t)gid * 16 + k] = sqrtf(best[k + 1]);
}

// ---------------- candidate matching: top24(bc) -> top8(xyz) -> top4(scf) ----------------
__global__ void match_kernel(const float* __restrict__ tbc, const float* __restrict__ sbc,
                             const float* __restrict__ txyz, const float* __restrict__ sxyz,
                             const float* __restrict__ sd, const float* __restrict__ td,
                             int* __restrict__ idxs)
{
  int gid = blockIdx.x * blockDim.x + threadIdx.x;
  if (gid >= BB * NN) return;
  int b = gid >> 10;
  int n = gid & (NN - 1);
  float q[9];
#pragma unroll
  for (int c = 0; c < 9; ++c) q[c] = sbc[((size_t)b * NN + n) * 9 + c];
  float bv[24]; int bix[24];
#pragma unroll
  for (int i = 0; i < 24; ++i) { bv[i] = 3.0e38f; bix[i] = 0; }
  for (int m = 0; m < MM; ++m) {
    const float* t = tbc + ((size_t)b * MM + m) * 9;
    float d2 = 0.f;
#pragma unroll
    for (int c = 0; c < 9; ++c) { float df = q[c] - t[c]; d2 += df * df; }
    if (d2 < bv[23]) {
      int p = 23;
      while (p > 0 && bv[p - 1] > d2) { bv[p] = bv[p - 1]; bix[p] = bix[p - 1]; --p; }
      bv[p] = d2; bix[p] = m;
    }
  }
  float px = sxyz[((size_t)b * NN + n) * 3 + 0];
  float py = sxyz[((size_t)b * NN + n) * 3 + 1];
  float pz = sxyz[((size_t)b * NN + n) * 3 + 2];
  float xv[8]; int xix[8];
#pragma unroll
  for (int i = 0; i < 8; ++i) { xv[i] = 3.0e38f; xix[i] = 0x7fffffff; }
  for (int c = 0; c < 24; ++c) {
    int m = bix[c];
    float dx = px - txyz[((size_t)b * MM + m) * 3 + 0];
    float dy = py - txyz[((size_t)b * MM + m) * 3 + 1];
    float dz = pz - txyz[((size_t)b * MM + m) * 3 + 2];
    float d2 = dx * dx + dy * dy + dz * dz;
    if ((d2 < xv[7]) || (d2 == xv[7] && m < xix[7])) {
      int p = 7;
      while (p > 0 && (xv[p - 1] > d2 || (xv[p - 1] == d2 && xix[p - 1] > m))) {
        xv[p] = xv[p - 1]; xix[p] = xix[p - 1]; --p;
      }
      xv[p] = d2; xix[p] = m;
    }
  }
  float md[16];
#pragma unroll
  for (int k = 0; k < 16; ++k) md[k] = sd[((size_t)b * NN + n) * 16 + k];
  float sv[4]; int six[4];
#pragma unroll
  for (int i = 0; i < 4; ++i) { sv[i] = 3.0e38f; six[i] = 0x7fffffff; }
  for (int c = 0; c < 8; ++c) {
    int m = xix[c];
    const float* tp = td + ((size_t)b * MM + m) * 16;
    float d2 = 0.f;
#pragma unroll
    for (int k = 0; k < 16; ++k) { float df = md[k] - tp[k]; d2 += df * df; }
    if ((d2 < sv[3]) || (d2 == sv[3] && m < six[3])) {
      int p = 3;
      while (p > 0 && (sv[p - 1] > d2 || (sv[p - 1] == d2 && six[p - 1] > m))) {
        sv[p] = sv[p - 1]; six[p] = six[p - 1]; --p;
      }
      sv[p] = d2; six[p] = m;
    }
  }
#pragma unroll
  for (int r = 0; r < 4; ++r) idxs[(size_t)gid * 4 + r] = six[r];
}

// ---------------- build MLP input rows (feature(256)+scf(3) pad->288) + pos(12) ----------------
__global__ void build_mlp_input_kernel(
    const float* __restrict__ tfeat, const float* __restrict__ sfeat,
    const float* __restrict__ txyz, const float* __restrict__ sxyz,
    const float* __restrict__ tbc, const float* __restrict__ sbc,
    const float* __restrict__ tscf, const float* __restrict__ sscf,
    const int* __restrict__ idxs,
    unsigned short* __restrict__ X, float* __restrict__ pos)
{
  int g = blockIdx.x;
  int b = g >> 10;
  int n = g & (NN - 1);
  int f = threadIdx.x;
  for (int s = 0; s < 5; ++s) {
    size_t row = (size_t)g * 5 + s;
    int m; const float* featp; const float* scfp; int Np;
    if (s == 0) { m = n; featp = sfeat + (size_t)b * FF * NN; scfp = sscf + (size_t)b * 3 * NN; Np = NN; }
    else { m = idxs[(size_t)g * 4 + (s - 1)]; featp = tfeat + (size_t)b * FF * MM; scfp = tscf + (size_t)b * 3 * MM; Np = MM; }
    X[row * 288 + f] = f2bf(featp[(size_t)f * Np + m]);
    if (f < 32) {
      int c = 256 + f;
      float v = (c < 259) ? scfp[(size_t)(c - 256) * Np + m] : 0.0f;
      X[row * 288 + c] = f2bf(v);
    }
    if (f < 12) {
      float v;
      if (f < 3) v = (s == 0) ? sxyz[((size_t)b * NN + n) * 3 + f] : txyz[((size_t)b * MM + m) * 3 + f];
      else {
        int c = f - 3;
        v = (s == 0) ? sbc[((size_t)b * NN + n) * 9 + c] : tbc[((size_t)b * MM + m) * 9 + c];
      }
      pos[row * 12 + f] = v;
    }
  }
}

// ---------------- rel_pos rows (i=0 only): RP[g*5+j, 0..31] ----------------
__global__ void build_relpos_kernel(const float* __restrict__ pos, unsigned short* __restrict__ RP)
{
  int gid = blockIdx.x * blockDim.x + threadIdx.x;
  if (gid >= R5 * 32) return;
  int row = gid >> 5;
  int c = gid & 31;
  int g = row / 5;
  float v = 0.f;
  if (c < 12) v = pos[((size_t)g * 5) * 12 + c] - pos[(size_t)row * 12 + c];
  RP[gid] = f2bf(v);
}

// ---------------- sim_in = q0 - k_j + pe ----------------
__global__ void build_simin_kernel(const unsigned short* __restrict__ qkvH,
                                   const float* __restrict__ peF,
                                   unsigned short* __restrict__ siminH)
{
  int gid = blockIdx.x * blockDim.x + threadIdx.x;
  if (gid >= R5 * 256) return;
  int row = gid >> 8;
  int d = gid & 255;
  int g = row / 5;
  float qv = bf2f(qkvH[((size_t)g * 5) * 768 + d]);
  float kv = bf2f(qkvH[(size_t)row * 768 + 256 + d]);
  siminH[gid] = f2bf(qv - kv + peF[gid]);
}

// ---------------- softmax over j + aggregate + residual search_feature ----------------
__global__ void softmax_agg_kernel(const float* __restrict__ sim,
                                   const unsigned short* __restrict__ qkvH,
                                   const float* __restrict__ peF,
                                   const float* __restrict__ sfeat,
                                   float* __restrict__ ffF)
{
  int gid = blockIdx.x * blockDim.x + threadIdx.x;
  if (gid >= GN * 256) return;
  int g = gid >> 8;
  int d = gid & 255;
  size_t base = (size_t)g * 5;
  float s[5];
  float mx = -3.0e38f;
#pragma unroll
  for (int j = 0; j < 5; ++j) { s[j] = sim[(base + j) * 256 + d]; mx = fmaxf(mx, s[j]); }
  float den = 0.f;
#pragma unroll
  for (int j = 0; j < 5; ++j) { s[j] = expf(s[j] - mx); den += s[j]; }
  float inv = 1.0f / den;
  float agg = 0.f;
#pragma unroll
  for (int j = 0; j < 5; ++j) {
    float vv = bf2f(qkvH[(base + j) * 768 + 512 + d]);
    float pe = peF[(base + j) * 256 + d];
    agg += s[j] * inv * (vv + pe);
  }
  int b = g >> 10;
  int n = g & (NN - 1);
  ffF[(size_t)g * 256 + d] = agg + sfeat[(size_t)b * FF * NN + (size_t)d * NN + n];
}

// ---------------- instance norm over channels (row of 256) ----------------
__global__ void inorm_kernel(const float* __restrict__ in, float* __restrict__ outF,
                             unsigned short* __restrict__ outH)
{
  int row = blockIdx.x;
  int tid = threadIdx.x;
  __shared__ float red[256];
  __shared__ float stats[2];
  float x = in[(size_t)row * 256 + tid];
  red[tid] = x;
  __syncthreads();
  for (int s = 128; s > 0; s >>= 1) { if (tid < s) red[tid] += red[tid + s]; __syncthreads(); }
  if (tid == 0) stats[0] = red[0] * (1.0f / 256.0f);
  __syncthreads();
  float mu = stats[0];
  float dv = x - mu;
  red[tid] = dv * dv;
  __syncthreads();
  for (int s = 128; s > 0; s >>= 1) { if (tid < s) red[tid] += red[tid + s]; __syncthreads(); }
  if (tid == 0) stats[1] = red[0] * (1.0f / 256.0f);
  __syncthreads();
  float y = dv * rsqrtf(stats[1] + 1e-5f);
  outF[(size_t)row * 256 + tid] = y;
  if (outH) outH[(size_t)row * 256 + tid] = f2bf(y);
}

// ---------------- PointSIFT: nearest neighbor per octant within radius ----------------
__global__ void oct_nn_kernel(const float* __restrict__ sxyz, int* __restrict__ idx8)
{
  int gid = blockIdx.x * blockDim.x + threadIdx.x;
  if (gid >= GN) return;
  int b = gid >> 10;
  int n = gid & (NN - 1);
  const float* base = sxyz + (size_t)b * NN * 3;
  float px = base[n * 3], py = base[n * 3 + 1], pz = base[n * 3 + 2];
  float mind[8]; int mi[8];
#pragma unroll
  for (int c = 0; c < 8; ++c) { mind[c] = 3.0e38f; mi[c] = n; }
  for (int j = 0; j < NN; ++j) {
    float dx = base[j * 3] - px, dy = base[j * 3 + 1] - py, dz = base[j * 3 + 2] - pz;
    float d2 = dx * dx + dy * dy + dz * dz;
    if (d2 > 0.25f || d2 <= 0.f) continue;   // RADIUS^2 = 0.25
    int oc = ((dx > 0.f) ? 4 : 0) + ((dy > 0.f) ? 2 : 0) + ((dz > 0.f) ? 1 : 0);
    if (d2 < mind[oc]) { mind[oc] = d2; mi[oc] = j; }
  }
#pragma unroll
  for (int c = 0; c < 8; ++c) idx8[(size_t)gid * 8 + c] = mi[c];
}

// ---------------- build PointSIFT conv(1,2) layer-1 input (K = 259*2 pad 544) ----------------
__global__ void build_oe_input_kernel(const float* __restrict__ feat,
                                      const float* __restrict__ sxyz,
                                      const int* __restrict__ idx8,
                                      unsigned short* __restrict__ A1)
{
  int g = blockIdx.x;
  int f = threadIdx.x;
  int b = g >> 10;
  int n = g & (NN - 1);
  const float* xyzb = sxyz + (size_t)b * NN * 3;
  for (int oct = 0; oct < 8; ++oct) {
    int j = idx8[(size_t)g * 8 + oct];
    size_t row = (size_t)g * 4 + (oct >> 1);
    int t = oct & 1;
    A1[row * 544 + (size_t)(3 + f) * 2 + t] = f2bf(feat[((size_t)b * NN + j) * 256 + f]);
    if (f < 3) {
      float d = xyzb[j * 3 + f] - xyzb[n * 3 + f];
      A1[row * 544 + (size_t)f * 2 + t] = f2bf(d);
    }
  }
  if (f < 104) {                       // zero pad K = 518..543 for the 4 rows
    int r = f / 26;
    int k = 518 + (f % 26);
    A1[((size_t)g * 4 + r) * 544 + k] = 0;
  }
}

// ---------------- repack conv output pairs: A[row,2c+t] = C[2*row+t, c] ----------------
__global__ void repack_pairs_kernel(const float* __restrict__ Cin,
                                    unsigned short* __restrict__ Aout, int rows)
{
  int gid = blockIdx.x * blockDim.x + threadIdx.x;
  if (gid >= rows * 256) return;
  int row = gid >> 8;
  int c = gid & 255;
  Aout[(size_t)row * 512 + 2 * c]     = f2bf(Cin[((size_t)row * 2 + 0) * 256 + c]);
  Aout[(size_t)row * 512 + 2 * c + 1] = f2bf(Cin[((size_t)row * 2 + 1) * 256 + c]);
}

// ---------------- final (B*N,F) rows -> (B,F,N) ----------------
__global__ void out_transpose_kernel(const float* __restrict__ yrows, float* __restrict__ out)
{
  int gid = blockIdx.x * blockDim.x + threadIdx.x;
  if (gid >= BB * FF * NN) return;
  int n = gid & (NN - 1);
  int rest = gid >> 10;
  int f = rest & 255;
  int b = rest >> 8;
  out[gid] = yrows[(((size_t)b << 10) + n) * 256 + f];
}

// ============================================================
extern "C" void kernel_launch(void* const* d_in, const int* in_sizes, int n_in,
                              void* d_out, int out_size, void* d_ws, size_t ws_size,
                              hipStream_t stream)
{
  (void)in_sizes; (void)n_in; (void)out_size; (void)ws_size;
  const float* tfeat = (const float*)d_in[0];
  const float* sfeat = (const float*)d_in[1];
  const float* txyz  = (const float*)d_in[2];
  const float* sxyz  = (const float*)d_in[3];
  const float* tbc   = (const float*)d_in[4];
  const float* sbc   = (const float*)d_in[5];
  const float* mlp_w1 = (const float*)d_in[6];
  const float* mlp_g1 = (const float*)d_in[7];
  const float* mlp_b1 = (const float*)d_in[8];
  const float* mlp_w2 = (const float*)d_in[9];
  const float* mlp_g2 = (const float*)d_in[10];
  const float* mlp_b2 = (const float*)d_in[11];
  const float* mlp_w3 = (const float*)d_in[12];
  const float* mlp_g3 = (const float*)d_in[13];
  const float* mlp_b3 = (const float*)d_in[14];
  const float* fea_w1 = (const float*)d_in[15];
  const float* fea_g1 = (const float*)d_in[16];
  const float* fea_b1 = (const float*)d_in[17];
  const float* fea_w2 = (const float*)d_in[18];
  const float* fea_b2 = (const float*)d_in[19];
  const float* attn_wqkv = (const float*)d_in[20];
  const float* pos_w1 = (const float*)d_in[21];
  const float* pos_b1 = (const float*)d_in[22];
  const float* pos_w2 = (const float*)d_in[23];
  const float* pos_b2 = (const float*)d_in[24];
  const float* amlp_w1 = (const float*)d_in[25];
  const float* amlp_b1 = (const float*)d_in[26];
  const float* amlp_w2 = (const float*)d_in[27];
  const float* amlp_b2 = (const float*)d_in[28];
  const float* oeW[2][3] = { { (const float*)d_in[29], (const float*)d_in[32], (const float*)d_in[35] },
                             { (const float*)d_in[38], (const float*)d_in[41], (const float*)d_in[44] } };
  const float* oeG[2][3] = { { (const float*)d_in[30], (const float*)d_in[33], (const float*)d_in[36] },
                             { (const float*)d_in[39], (const float*)d_in[42], (const float*)d_in[45] } };
  const float* oeB[2][3] = { { (const float*)d_in[31], (const float*)d_in[34], (const float*)d_in[37] },
                             { (const float*)d_in[40], (const float*)d_in[43], (const float*)d_in[46] } };

  // ---------------- workspace layout ----------------
  char* wsb = (char*)d_ws;
  size_t off = 0;
  auto alloc = [&](size_t bytes) -> void* {
    void* p = (void*)(wsb + off);
    off = (off + bytes + 255) & ~(size_t)255;
    return p;
  };
  float* tscf = (float*)alloc((size_t)BB * 3 * MM * 4);
  float* sscf = (float*)alloc((size_t)BB * 3 * NN * 4);
  float* td   = (float*)alloc((size_t)BB * MM * 16 * 4);
  float* sd   = (float*)alloc((size_t)BB * NN * 16 * 4);
  int*   idxs = (int*)alloc((size_t)GN * 4 * 4);
  unsigned short* mlp_w1p = (unsigned short*)alloc(256 * 288 * 2);
  unsigned short* mlp_w2p = (unsigned short*)alloc(256 * 256 * 2);
  unsigned short* mlp_w3p = (unsigned short*)alloc(256 * 256 * 2);
  unsigned short* fea_w1p = (unsigned short*)alloc(256 * 256 * 2);
  unsigned short* fea_w2p = (unsigned short*)alloc(256 * 256 * 2);
  unsigned short* qkvWp   = (unsigned short*)alloc(768 * 256 * 2);
  unsigned short* posw1p  = (unsigned short*)alloc(64 * 32 * 2);
  unsigned short* posw2p  = (unsigned short*)alloc(256 * 64 * 2);
  unsigned short* aw1p    = (unsigned short*)alloc(1024 * 256 * 2);
  unsigned short* aw2p    = (unsigned short*)alloc((size_t)256 * 1024 * 2);
  unsigned short* oeWp[2][3];
  for (int md = 0; md < 2; ++md) {
    oeWp[md][0] = (unsigned short*)alloc(256 * 544 * 2);
    oeWp[md][1] = (unsigned short*)alloc(256 * 512 * 2);
    oeWp[md][2] = (unsigned short*)alloc(256 * 512 * 2);
  }
  unsigned short* Xmlp   = (unsigned short*)alloc((size_t)R5 * 288 * 2);
  unsigned short* hA     = (unsigned short*)alloc((size_t)R5 * 256 * 2);
  unsigned short* hB     = (unsigned short*)alloc((size_t)R5 * 256 * 2);
  float*          pos    = (float*)alloc((size_t)R5 * 12 * 4);
  unsigned short* qkvH   = (unsigned short*)alloc((size_t)R5 * 768 * 2);
  unsigned short* RP     = (unsigned short*)alloc((size_t)R5 * 32 * 2);
  unsigned short* peH    = (unsigned short*)alloc((size_t)R5 * 64 * 2);
  float*          peF    = (float*)alloc((size_t)R5 * 256 * 4);
  unsigned short* siminH = (unsigned short*)alloc((size_t)R5 * 256 * 2);
  unsigned short* hhH    = (unsigned short*)alloc((size_t)R5 * 1024 * 2);
  float*          simF   = (float*)alloc((size_t)R5 * 256 * 4);
  float*          ffF    = (float*)alloc((size_t)GN * 256 * 4);
  float*          ffnF   = (float*)alloc((size_t)GN * 256 * 4);
  unsigned short* ffnH   = (unsigned short*)alloc((size_t)GN * 256 * 2);
  unsigned short* y1H    = (unsigned short*)alloc((size_t)GN * 256 * 2);
  float*          tmpF   = (float*)alloc((size_t)GN * 256 * 4);
  float*          ynF    = (float*)alloc((size_t)GN * 256 * 4);
  int*            idx8   = (int*)alloc((size_t)GN * 8 * 4);
  unsigned short* A1     = (unsigned short*)alloc((size_t)GN * 4 * 544 * 2);
  float*          C1     = (float*)alloc((size_t)GN * 4 * 256 * 4);
  unsigned short* A2     = (unsigned short*)alloc((size_t)GN * 2 * 512 * 2);
  float*          C2     = (float*)alloc((size_t)GN * 2 * 256 * 4);
  unsigned short* A3     = (unsigned short*)alloc((size_t)GN * 512 * 2);
  float*          y2F    = (float*)alloc((size_t)GN * 256 * 4);
  float*          y3F    = (float*)alloc((size_t)GN * 256 * 4);

  auto gemm = [&](const unsigned short* A, const unsigned short* Wp,
                  float* oF, unsigned short* oH,
                  const float* sc, const float* bi, const float* res,
                  int Mr, int Oc, int Kp, int relu) {
    dim3 grid((Mr / 16 + 7) / 8, Oc / 16);
    size_t shmem = (size_t)16 * Kp * 2;     // 16-row weight strip in LDS
    gemm_bf16_kernel<<<grid, 256, shmem, stream>>>(A, Wp, oF, oH, sc, bi, res, Mr, Oc, Kp, relu);
  };
  auto pack = [&](const float* Wsrc, unsigned short* dst, int O, int K, int Kp) {
    int total = O * Kp;
    pack_w_kernel<<<(total + 255) / 256, 256, 0, stream>>>(Wsrc, dst, O, K, Kp, total);
  };

  // ---- pack all weights to padded bf16 ----
  pack(mlp_w1, mlp_w1p, 256, 259, 288);
  pack(mlp_w2, mlp_w2p, 256, 256, 256);
  pack(mlp_w3, mlp_w3p, 256, 256, 256);
  pack(fea_w1, fea_w1p, 256, 256, 256);
  pack(fea_w2, fea_w2p, 256, 256, 256);
  pack(attn_wqkv, qkvWp, 768, 256, 256);
  pack(pos_w1, posw1p, 64, 12, 32);
  pack(pos_w2, posw2p, 256, 64, 64);
  pack(amlp_w1, aw1p, 1024, 256, 256);
  pack(amlp_w2, aw2p, 256, 1024, 1024);
  for (int md = 0; md < 2; ++md) {
    pack(oeW[md][0], oeWp[md][0], 256, 518, 544);
    pack(oeW[md][1], oeWp[md][1], 256, 512, 512);
    pack(oeW[md][2], oeWp[md][2], 256, 512, 512);
  }

  // ---- geometry / matching ----
  scf_kernel<<<BB, 256, 0, stream>>>(txyz, tscf, MM);
  scf_kernel<<<BB, 256, 0, stream>>>(sxyz, sscf, NN);
  knn_desc_kernel<<<(BB * MM + 255) / 256, 256, 0, stream>>>(txyz, td, MM, BB * MM);
  knn_desc_kernel<<<(BB * NN + 255) / 256, 256, 0, stream>>>(sxyz, sd, NN, BB * NN);
  match_kernel<<<(BB * NN + 255) / 256, 256, 0, stream>>>(tbc, sbc, txyz, sxyz, sd, td, idxs);

  // ---- grouping + shared MLP (WMMA) ----
  build_mlp_input_kernel<<<GN, 256, 0, stream>>>(tfeat, sfeat, txyz, sxyz, tbc, sbc,
                                                 tscf, sscf, idxs, Xmlp, pos);
  gemm(Xmlp, mlp_w1p, nullptr, hA, mlp_g1, mlp_b1, nullptr, R5, 256, 288, 1);
  gemm(hA,   mlp_w2p, nullptr, hB, mlp_g2, mlp_b2, nullptr, R5, 256, 256, 1);
  gemm(hB,   mlp_w3p, nullptr, hA, mlp_g3, mlp_b3, nullptr, R5, 256, 256, 1);

  // ---- point transformer (only i=0 query row contributes to output) ----
  gemm(hA, qkvWp, nullptr, qkvH, nullptr, nullptr, nullptr, R5, 768, 256, 0);
  build_relpos_kernel<<<(R5 * 32 + 255) / 256, 256, 0, stream>>>(pos, RP);
  gemm(RP,  posw1p, nullptr, peH, nullptr, pos_b1, nullptr, R5, 64, 32, 1);
  gemm(peH, posw2p, peF, nullptr, nullptr, pos_b2, nullptr, R5, 256, 64, 0);
  build_simin_kernel<<<(R5 * 256 + 255) / 256, 256, 0, stream>>>(qkvH, peF, siminH);
  gemm(siminH, aw1p, nullptr, hhH, nullptr, amlp_b1, nullptr, R5, 1024, 256, 1);
  gemm(hhH,    aw2p, simF, nullptr, nullptr, amlp_b2, nullptr, R5, 256, 1024, 0);
  softmax_agg_kernel<<<(GN * 256 + 255) / 256, 256, 0, stream>>>(simF, qkvH, peF, sfeat, ffF);

  // ---- residual + instance norms + fea layer ----
  inorm_kernel<<<GN, 256, 0, stream>>>(ffF, ffnF, ffnH);
  gemm(ffnH, fea_w1p, nullptr, y1H, fea_g1, fea_b1, nullptr, GN, 256, 256, 1);
  gemm(y1H,  fea_w2p, tmpF, nullptr, nullptr, fea_b2, ffnF, GN, 256, 256, 0);
  inorm_kernel<<<GN, 256, 0, stream>>>(tmpF, ynF, nullptr);

  // ---- two PointSIFT OE modules with residuals ----
  oct_nn_kernel<<<(GN + 255) / 256, 256, 0, stream>>>(sxyz, idx8);
  for (int md = 0; md < 2; ++md) {
    const float* yin = (md == 0) ? ynF : y2F;
    float* yout = (md == 0) ? y2F : y3F;
    build_oe_input_kernel<<<GN, 256, 0, stream>>>(yin, sxyz, idx8, A1);
    gemm(A1, oeWp[md][0], C1, nullptr, oeG[md][0], oeB[md][0], nullptr, GN * 4, 256, 544, 1);
    repack_pairs_kernel<<<(GN * 2 * 256 + 255) / 256, 256, 0, stream>>>(C1, A2, GN * 2);
    gemm(A2, oeWp[md][1], C2, nullptr, oeG[md][1], oeB[md][1], nullptr, GN * 2, 256, 512, 1);
    repack_pairs_kernel<<<(GN * 256 + 255) / 256, 256, 0, stream>>>(C2, A3, GN);
    gemm(A3, oeWp[md][2], yout, nullptr, oeG[md][2], oeB[md][2], yin, GN, 256, 512, 1);
  }

  // ---- (B*N,F) rows -> (B,F,N) output ----
  out_transpose_kernel<<<(BB * FF * NN + 255) / 256, 256, 0, stream>>>(y3F, (float*)d_out);
}
